// HivCausalGIN_46909632806969
// MI455X (gfx1250) — matrix operand order repaired and checked by reading, not compiled, not run on hardware
//
#include <hip/hip_runtime.h>
#include <math.h>

#define Nn 50000
#define Ee 600000
#define Gg 512
#define Dd 128
#define Ll 3

typedef __attribute__((ext_vector_type(16))) __bf16       v16bf;
typedef __attribute__((ext_vector_type(8)))  float        v8f;
typedef __attribute__((ext_vector_type(4)))  unsigned int u32x4;

union Frag {
  v16bf v;
  unsigned short u[16];
  u32x4 q[2];
};

__device__ __forceinline__ unsigned short f2bf(float f) {
  unsigned int x = __float_as_uint(f);
  x += 0x7fffu + ((x >> 16) & 1u);
  return (unsigned short)(x >> 16);
}

// ======================= WMMA GEMM =======================
// C[M,Nc] = A[M,K](bf16, row-major) @ W[K,Nc] + bias, with W supplied
// PRE-TRANSPOSED as Wt[Nc,K] so both fragments load as contiguous b128s.
// One wave computes one 16x16 tile; K templated so loop fully unrolls.
template <int K>
__global__ void wmma_gemm_k(const unsigned short* __restrict__ A,
                            const unsigned short* __restrict__ Wt,
                            const float* __restrict__ bias,
                            float* __restrict__ Cmat,
                            int M, int Nc, int relu) {
  int wave = blockIdx.x * (blockDim.x >> 5) + (threadIdx.x >> 5);
  int ntc = Nc >> 4;
  int tiles = (M >> 4) * ntc;
  if (wave >= tiles) return;
  int rt = wave / ntc;
  int ct = wave - rt * ntc;
  int lane = threadIdx.x & 31;
  int half = lane >> 4;
  int l16  = lane & 15;

  const unsigned short* Arow = A  + (rt * 16 + l16) * K;   // A row, K-contiguous
  int ncol = ct * 16 + l16;
  const unsigned short* Wrow = Wt + ncol * K;              // W column, K-contiguous

  v8f acc = {0.f, 0.f, 0.f, 0.f, 0.f, 0.f, 0.f, 0.f};
#pragma unroll
  for (int k0 = 0; k0 < K; k0 += 32) {
    // A frag (16x32): lanes 0-15 -> K = k0+0..7 / k0+16..23; lanes 16-31 -> +8.
    Frag a;
    a.q[0] = *reinterpret_cast<const u32x4*>(Arow + k0 + half * 8);
    a.q[1] = *reinterpret_cast<const u32x4*>(Arow + k0 + 16 + half * 8);
    // B frag (32x16): lane = column; lanes 0-15 -> K = k0+0..7 / k0+16..23; lanes 16-31 -> +8.
    Frag b;
    b.q[0] = *reinterpret_cast<const u32x4*>(Wrow + k0 + half * 8);
    b.q[1] = *reinterpret_cast<const u32x4*>(Wrow + k0 + 16 + half * 8);
    acc = __builtin_amdgcn_wmma_f32_16x16x32_bf16(false, a.v, false, b.v,
                                                  (short)0, acc, false, false);
  }
  float bv = bias ? bias[ncol] : 0.0f;
  float* Crow = Cmat + (size_t)(rt * 16 + half * 8) * Nc + ncol;
#pragma unroll
  for (int r = 0; r < 8; ++r) {
    float vv = acc[r] + bv;
    if (relu) vv = fmaxf(vv, 0.0f);
    Crow[(size_t)r * Nc] = vv;
  }
}

// ======================= elementwise / util =======================
__global__ void to_bf16_k(const float* __restrict__ in, unsigned short* __restrict__ out, int n) {
  int i = blockIdx.x * blockDim.x + threadIdx.x;
  if (i < n) out[i] = f2bf(in[i]);
}

// transpose + convert: wt[n*K + k] = bf16(w[k*Nc + n])
__global__ void wt_bf16_k(const float* __restrict__ w, unsigned short* __restrict__ wt,
                          int K, int Nc) {
  int i = blockIdx.x * blockDim.x + threadIdx.x;
  if (i >= K * Nc) return;
  int k = i / Nc;
  int n = i - k * Nc;
  wt[n * K + k] = f2bf(w[i]);
}

__global__ void copyf_k(const float* __restrict__ in, float* __restrict__ out, int n) {
  int i = blockIdx.x * blockDim.x + threadIdx.x;
  if (i < n) out[i] = in[i];
}

__global__ void fillf_k(float* __restrict__ p, float v, int n) {
  int i = blockIdx.x * blockDim.x + threadIdx.x;
  if (i < n) p[i] = v;
}

// ======================= embeddings =======================
__global__ void node_embed_k(const int* __restrict__ xf, const float* __restrict__ atab,
                             float* __restrict__ h) {
  int n = blockIdx.x;
  int d = threadIdx.x;
  float s = 0.f;
#pragma unroll
  for (int f = 0; f < 9; ++f) {
    int idx = xf[n * 9 + f];
    s += atab[((size_t)f * 119 + idx) * Dd + d];
  }
  h[(size_t)n * Dd + d] = s;
}

__global__ void vn_init_k(const float* __restrict__ vne, float* __restrict__ vn) {
  vn[(size_t)blockIdx.x * Dd + threadIdx.x] = vne[threadIdx.x];
}

__global__ void add_vn_k(float* __restrict__ h, const float* __restrict__ vn,
                         const int* __restrict__ batch, int n) {
  int i = blockIdx.x * blockDim.x + threadIdx.x;
  if (i >= n) return;
  int node = i >> 7;
  int d = i & 127;
  h[i] += vn[(size_t)batch[node] * Dd + d];
}

// ======================= edge message + aggregate =======================
__global__ void edge_msg_k(const int* __restrict__ eattr, const int* __restrict__ src,
                           const int* __restrict__ dst, const float* __restrict__ btab,
                           const float* __restrict__ h, float* __restrict__ agg, int ne) {
  int e = blockIdx.x * (blockDim.x >> 5) + (threadIdx.x >> 5);
  if (e >= ne) return;
  int lane = threadIdx.x & 31;
  int a0 = eattr[e * 3 + 0], a1 = eattr[e * 3 + 1], a2 = eattr[e * 3 + 2];
  int s = src[e], d = dst[e];
  const float* hr = h + (size_t)s * Dd;
  float* ar = agg + (size_t)d * Dd;
#pragma unroll
  for (int j = 0; j < 4; ++j) {
    int k = lane + j * 32;
    float ev = btab[(0 * 6 + a0) * Dd + k] + btab[(1 * 6 + a1) * Dd + k] +
               btab[(2 * 6 + a2) * Dd + k];
    float m = fmaxf(hr[k] + ev, 0.f);
    atomicAdd(&ar[k], m);
  }
}

__global__ void combine_k(const float* __restrict__ h, const float* __restrict__ agg,
                          const float* __restrict__ eps, int l,
                          unsigned short* __restrict__ out, int n) {
  int i = blockIdx.x * blockDim.x + threadIdx.x;
  if (i >= n) return;
  float ep = 1.0f + eps[l];
  out[i] = f2bf(ep * h[i] + agg[i]);
}

// ======================= BatchNorm =======================
__global__ void bn_stats_k(const float* __restrict__ x, float* __restrict__ s,
                           int M, int C, int rpb) {
  int c = threadIdx.x;
  int r0 = blockIdx.x * rpb;
  int r1 = r0 + rpb; if (r1 > M) r1 = M;
  float a = 0.f, b = 0.f;
  for (int r = r0; r < r1; ++r) {
    float v = x[(size_t)r * C + c];
    a += v; b += v * v;
  }
  atomicAdd(&s[c], a);
  atomicAdd(&s[C + c], b);
}

__global__ void bn_apply_k(const float* __restrict__ x, const float* __restrict__ s,
                           const float* __restrict__ g, const float* __restrict__ b,
                           float* __restrict__ outf, unsigned short* __restrict__ outb,
                           int M, int C, int relu) {
  int i = blockIdx.x * blockDim.x + threadIdx.x;
  if (i >= M * C) return;
  int c = i % C;
  float invM = 1.0f / (float)M;
  float mean = s[c] * invM;
  float var = s[C + c] * invM - mean * mean;
  float inv = rsqrtf(var + 1e-5f);
  float v = (x[i] - mean) * inv * g[c] + b[c];
  if (relu) v = fmaxf(v, 0.f);
  if (outf) outf[i] = v;
  if (outb) outb[i] = f2bf(v);
}

// ======================= VN segment sum =======================
__global__ void seg_add_k(const float* __restrict__ h, const int* __restrict__ batch,
                          float* __restrict__ vt, int n) {
  int i = blockIdx.x * blockDim.x + threadIdx.x;
  if (i >= n) return;
  int node = i >> 7;
  int d = i & 127;
  atomicAdd(&vt[(size_t)batch[node] * Dd + d], h[i]);
}

// ======================= attention heads =======================
__global__ void edge_att_k(const float* __restrict__ x, const int* __restrict__ src,
                           const int* __restrict__ dst, const float* __restrict__ w,
                           const float* __restrict__ b, float* __restrict__ ew, int ne) {
  int e = blockIdx.x * (blockDim.x >> 5) + (threadIdx.x >> 5);
  if (e >= ne) return;
  int lane = threadIdx.x & 31;
  const float* xs = x + (size_t)src[e] * Dd;
  const float* xd = x + (size_t)dst[e] * Dd;
  float s0 = 0.f, s1 = 0.f;
  for (int k = lane; k < Dd; k += 32) {
    float v = xs[k];
    s0 += v * w[k * 2 + 0]; s1 += v * w[k * 2 + 1];
  }
  for (int k = lane; k < Dd; k += 32) {
    float v = xd[k];
    s0 += v * w[(Dd + k) * 2 + 0]; s1 += v * w[(Dd + k) * 2 + 1];
  }
  for (int o = 16; o > 0; o >>= 1) { s0 += __shfl_xor(s0, o, 32); s1 += __shfl_xor(s1, o, 32); }
  if (lane == 0) {
    s0 += b[0]; s1 += b[1];
    float m = fmaxf(s0, s1);
    float e0 = __expf(s0 - m), e1 = __expf(s1 - m);
    ew[e] = e1 / (e0 + e1);
  }
}

__global__ void node_att_k(const float* __restrict__ x, const float* __restrict__ w,
                           const float* __restrict__ b, float* __restrict__ xo, int n) {
  int node = blockIdx.x * (blockDim.x >> 5) + (threadIdx.x >> 5);
  if (node >= n) return;
  int lane = threadIdx.x & 31;
  const float* xr = x + (size_t)node * Dd;
  float s0 = 0.f, s1 = 0.f;
  for (int k = lane; k < Dd; k += 32) {
    float v = xr[k];
    s0 += v * w[k * 2 + 0]; s1 += v * w[k * 2 + 1];
  }
  for (int o = 16; o > 0; o >>= 1) { s0 += __shfl_xor(s0, o, 32); s1 += __shfl_xor(s1, o, 32); }
  s0 += b[0]; s1 += b[1];
  float m = fmaxf(s0, s1);
  float e0 = __expf(s0 - m), e1 = __expf(s1 - m);
  float att = e1 / (e0 + e1);
  for (int k = lane; k < Dd; k += 32) xo[(size_t)node * Dd + k] = att * xr[k];
}

// ======================= GCN propagation =======================
__global__ void deg_edges_k(const int* __restrict__ src, const float* __restrict__ ew,
                            float* __restrict__ deg, int ne) {
  int i = blockIdx.x * blockDim.x + threadIdx.x;
  if (i < ne) atomicAdd(&deg[src[i]], ew[i]);
}

__global__ void dinv_k(const float* __restrict__ deg, float* __restrict__ dinv, int n) {
  int i = blockIdx.x * blockDim.x + threadIdx.x;
  if (i < n) dinv[i] = (deg[i] > 0.f) ? rsqrtf(deg[i]) : 0.f;
}

__global__ void self_loop_k(const float* __restrict__ dinv, const float* __restrict__ xw,
                            float* __restrict__ outp, int n) {
  int i = blockIdx.x * blockDim.x + threadIdx.x;
  if (i >= n) return;
  int node = i >> 7;
  float dv = dinv[node];
  outp[i] = dv * dv * xw[i];
}

__global__ void edge_gcn_k(const int* __restrict__ src, const int* __restrict__ dst,
                           const float* __restrict__ ew, const float* __restrict__ dinv,
                           const float* __restrict__ xw, float* __restrict__ outp, int ne) {
  int e = blockIdx.x * (blockDim.x >> 5) + (threadIdx.x >> 5);
  if (e >= ne) return;
  int lane = threadIdx.x & 31;
  int s = src[e], d = dst[e];
  float coef = dinv[s] * ew[e] * dinv[d];
  const float* xr = xw + (size_t)s * Dd;
  float* orow = outp + (size_t)d * Dd;
#pragma unroll
  for (int j = 0; j < 4; ++j) {
    int k = lane + j * 32;
    atomicAdd(&orow[k], coef * xr[k]);
  }
}

__global__ void relu_pool_k(const float* __restrict__ outp, const float* __restrict__ bias,
                            const int* __restrict__ batch, float* __restrict__ pooled, int n) {
  int i = blockIdx.x * blockDim.x + threadIdx.x;
  if (i >= n) return;
  int node = i >> 7;
  int d = i & 127;
  float v = fmaxf(outp[i] + bias[d], 0.f);
  atomicAdd(&pooled[(size_t)batch[node] * Dd + d], v);
}

// ======================= output head =======================
__global__ void head_k(const float* __restrict__ y, const float* __restrict__ w,
                       const float* __restrict__ b, float* __restrict__ outp) {
  int g = blockIdx.x * (blockDim.x >> 5) + (threadIdx.x >> 5);
  if (g >= Gg) return;
  int lane = threadIdx.x & 31;
  const float* yr = y + (size_t)g * Dd;
  float s0 = 0.f, s1 = 0.f;
  for (int k = lane; k < Dd; k += 32) {
    float v = yr[k];
    s0 += v * w[k * 2 + 0]; s1 += v * w[k * 2 + 1];
  }
  for (int o = 16; o > 0; o >>= 1) { s0 += __shfl_xor(s0, o, 32); s1 += __shfl_xor(s1, o, 32); }
  if (lane == 0) { outp[g * 2 + 0] = s0 + b[0]; outp[g * 2 + 1] = s1 + b[1]; }
}

// ======================= host orchestration =======================
extern "C" void kernel_launch(void* const* d_in, const int* in_sizes, int n_in,
                              void* d_out, int out_size, void* d_ws, size_t ws_size,
                              hipStream_t stream) {
  const int* x_feat    = (const int*)d_in[0];
  const int* edge_attr = (const int*)d_in[1];
  const int* edge_idx  = (const int*)d_in[2];
  const int* batch     = (const int*)d_in[3];
  const int* src = edge_idx;
  const int* dst = edge_idx + Ee;

  const float* atom_tab   = (const float*)d_in[4];
  const float* bond_tab   = (const float*)d_in[5];
  const float* vn_emb     = (const float*)d_in[6];
  const float* eps        = (const float*)d_in[7];
  const float* gin_w1     = (const float*)d_in[8];
  const float* gin_b1     = (const float*)d_in[9];
  const float* gin_bn1_g  = (const float*)d_in[10];
  const float* gin_bn1_b  = (const float*)d_in[11];
  const float* gin_w2     = (const float*)d_in[12];
  const float* gin_b2     = (const float*)d_in[13];
  const float* layer_bn_g = (const float*)d_in[14];
  const float* layer_bn_b = (const float*)d_in[15];
  const float* vn_w1      = (const float*)d_in[16];
  const float* vn_b1      = (const float*)d_in[17];
  const float* vn_bn1_g   = (const float*)d_in[18];
  const float* vn_bn1_b   = (const float*)d_in[19];
  const float* vn_w2      = (const float*)d_in[20];
  const float* vn_b2      = (const float*)d_in[21];
  const float* vn_bn2_g   = (const float*)d_in[22];
  const float* vn_bn2_b   = (const float*)d_in[23];
  const float* edge_att_w = (const float*)d_in[24];
  const float* edge_att_b = (const float*)d_in[25];
  const float* node_att_w = (const float*)d_in[26];
  const float* node_att_b = (const float*)d_in[27];
  const float* bno_g      = (const float*)d_in[28];
  const float* bno_b      = (const float*)d_in[29];
  const float* gcn_w      = (const float*)d_in[30];
  const float* gcn_b      = (const float*)d_in[31];
  const float* omlp_bn1_g = (const float*)d_in[32];
  const float* omlp_bn1_b = (const float*)d_in[33];
  const float* omlp_w1    = (const float*)d_in[34];
  const float* omlp_b1    = (const float*)d_in[35];
  const float* omlp_bn2_g = (const float*)d_in[36];
  const float* omlp_bn2_b = (const float*)d_in[37];
  const float* omlp_w2    = (const float*)d_in[38];
  const float* omlp_b2    = (const float*)d_in[39];

  float* yout = (float*)d_out;

  // ---- workspace carve-up ----
  char* wsp = (char*)d_ws;
  size_t off = 0;
  auto carve = [&](size_t bytes) -> void* {
    void* p = wsp + off;
    off += (bytes + 255) & ~(size_t)255;
    return p;
  };
  float* h    = (float*)carve((size_t)Nn * Dd * 4);
  float* agg  = (float*)carve((size_t)Nn * Dd * 4);          // reused as 'out'
  float* z2   = (float*)carve((size_t)Nn * Dd * 4);          // reused as 'xo'
  float* t1   = (float*)carve((size_t)Nn * 2 * Dd * 4);      // reused as 'xw'
  unsigned short* bfA = (unsigned short*)carve((size_t)Nn * 2 * Dd * 2);
  unsigned short* bfB = (unsigned short*)carve((size_t)Nn * 2 * Dd * 2);
  float* stats  = (float*)carve(512 * 4);
  float* vn     = (float*)carve((size_t)Gg * Dd * 4);
  float* vt     = (float*)carve((size_t)Gg * Dd * 4);        // reused as final 'y' f32
  float* t1v    = (float*)carve((size_t)Gg * 2 * Dd * 4);
  float* yv     = (float*)carve((size_t)Gg * Dd * 4);
  float* ew     = (float*)carve((size_t)Ee * 4);
  float* deg    = (float*)carve((size_t)Nn * 4);
  float* dinv   = (float*)carve((size_t)Nn * 4);
  float* pooled = (float*)carve((size_t)Gg * Dd * 4);
  unsigned short* wb = (unsigned short*)carve((size_t)400000 * 2);
  (void)ws_size; (void)n_in; (void)in_sizes; (void)out_size;

  // transposed bf16 weights: Wt[Nc x K]
  unsigned short* gin_w1t  = wb;                  // 3 * 256x128
  unsigned short* gin_w2t  = wb + 98304;          // 3 * 128x256
  unsigned short* vn_w1t   = wb + 196608;         // 2 * 256x128
  unsigned short* vn_w2t   = wb + 262144;         // 2 * 128x256
  unsigned short* gcn_wt   = wb + 327680;         // 128x128
  unsigned short* omlp_w1t = wb + 344064;         // 128x128

  const int ND = Nn * Dd;
  const int GD = Gg * Dd;

  auto cvt = [&](const float* in, unsigned short* out, int n) {
    to_bf16_k<<<(n + 255) / 256, 256, 0, stream>>>(in, out, n);
  };
  auto wtc = [&](const float* w, unsigned short* wt, int K, int Nc) {
    int n = K * Nc;
    wt_bf16_k<<<(n + 255) / 256, 256, 0, stream>>>(w, wt, K, Nc);
  };
  auto gemm = [&](const unsigned short* A, const unsigned short* Wt, const float* bias,
                  float* Cm, int M, int K, int Nc, int relu) {
    int tiles = (M / 16) * (Nc / 16);
    int blocks = (tiles + 3) / 4;   // 4 waves / 128 threads per block
    if (K == 128)
      wmma_gemm_k<128><<<blocks, 128, 0, stream>>>(A, Wt, bias, Cm, M, Nc, relu);
    else
      wmma_gemm_k<256><<<blocks, 128, 0, stream>>>(A, Wt, bias, Cm, M, Nc, relu);
  };
  auto bn = [&](const float* x, int M, int C, const float* g, const float* b,
                float* outf, unsigned short* outb, int relu) {
    hipMemsetAsync(stats, 0, 2 * C * sizeof(float), stream);
    int rpb = 128;
    bn_stats_k<<<(M + rpb - 1) / rpb, C, 0, stream>>>(x, stats, M, C, rpb);
    int tot = M * C;
    bn_apply_k<<<(tot + 255) / 256, 256, 0, stream>>>(x, stats, g, b, outf, outb, M, C, relu);
  };

  // ---- weight transpose+convert to bf16 (per-layer slabs) ----
  for (int l = 0; l < Ll; ++l) {
    wtc(gin_w1 + (size_t)l * 128 * 256, gin_w1t + (size_t)l * 256 * 128, 128, 256);
    wtc(gin_w2 + (size_t)l * 256 * 128, gin_w2t + (size_t)l * 128 * 256, 256, 128);
  }
  for (int l = 0; l < Ll - 1; ++l) {
    wtc(vn_w1 + (size_t)l * 128 * 256, vn_w1t + (size_t)l * 256 * 128, 128, 256);
    wtc(vn_w2 + (size_t)l * 256 * 128, vn_w2t + (size_t)l * 128 * 256, 256, 128);
  }
  wtc(gcn_w, gcn_wt, 128, 128);
  wtc(omlp_w1, omlp_w1t, 128, 128);

  // ---- node embedding + virtual-node init ----
  node_embed_k<<<Nn, Dd, 0, stream>>>(x_feat, atom_tab, h);
  vn_init_k<<<Gg, Dd, 0, stream>>>(vn_emb, vn);

  const int edgeBlocks = (Ee + 7) / 8;  // wave per edge, 8 waves/block

  // ---- GIN layers ----
  for (int l = 0; l < Ll; ++l) {
    add_vn_k<<<(ND + 255) / 256, 256, 0, stream>>>(h, vn, batch, ND);

    hipMemsetAsync(agg, 0, (size_t)ND * 4, stream);
    edge_msg_k<<<edgeBlocks, 256, 0, stream>>>(edge_attr, src, dst,
                                               bond_tab + (size_t)l * 3 * 6 * Dd, h, agg, Ee);

    combine_k<<<(ND + 255) / 256, 256, 0, stream>>>(h, agg, eps, l, bfB, ND);

    gemm(bfB, gin_w1t + (size_t)l * 256 * 128, gin_b1 + l * 256, t1, Nn, 128, 256, 0);
    bn(t1, Nn, 256, gin_bn1_g + l * 256, gin_bn1_b + l * 256, nullptr, bfA, 1);
    gemm(bfA, gin_w2t + (size_t)l * 128 * 256, gin_b2 + l * 128, z2, Nn, 256, 128, 0);

    if (l < Ll - 1) {
      copyf_k<<<(GD + 255) / 256, 256, 0, stream>>>(vn, vt, GD);
      seg_add_k<<<(ND + 255) / 256, 256, 0, stream>>>(h, batch, vt, ND);
    }

    bn(z2, Nn, 128, layer_bn_g + l * 128, layer_bn_b + l * 128, h, nullptr, (l < Ll - 1) ? 1 : 0);

    if (l < Ll - 1) {
      cvt(vt, bfB, GD);
      gemm(bfB, vn_w1t + (size_t)l * 256 * 128, vn_b1 + l * 256, t1v, Gg, 128, 256, 0);
      bn(t1v, Gg, 256, vn_bn1_g + l * 256, vn_bn1_b + l * 256, nullptr, bfA, 1);
      gemm(bfA, vn_w2t + (size_t)l * 128 * 256, vn_b2 + l * 128, yv, Gg, 256, 128, 0);
      bn(yv, Gg, 128, vn_bn2_g + l * 128, vn_bn2_b + l * 128, vn, nullptr, 1);
    }
  }

  // ---- edge / node attention ----
  edge_att_k<<<edgeBlocks, 256, 0, stream>>>(h, src, dst, edge_att_w, edge_att_b, ew, Ee);
  node_att_k<<<(Nn + 7) / 8, 256, 0, stream>>>(h, node_att_w, node_att_b, z2, Nn);

  bn(z2, Nn, 128, bno_g, bno_b, nullptr, bfA, 0);
  float* xw = t1;
  gemm(bfA, gcn_wt, nullptr, xw, Nn, 128, 128, 0);

  fillf_k<<<(Nn + 255) / 256, 256, 0, stream>>>(deg, 1.0f, Nn);
  deg_edges_k<<<(Ee + 255) / 256, 256, 0, stream>>>(src, ew, deg, Ee);
  dinv_k<<<(Nn + 255) / 256, 256, 0, stream>>>(deg, dinv, Nn);

  float* outb = agg;
  self_loop_k<<<(ND + 255) / 256, 256, 0, stream>>>(dinv, xw, outb, ND);
  edge_gcn_k<<<edgeBlocks, 256, 0, stream>>>(src, dst, ew, dinv, xw, outb, Ee);

  hipMemsetAsync(pooled, 0, (size_t)GD * 4, stream);
  relu_pool_k<<<(ND + 255) / 256, 256, 0, stream>>>(outb, gcn_b, batch, pooled, ND);

  // ---- output MLP ----
  bn(pooled, Gg, 128, omlp_bn1_g, omlp_bn1_b, nullptr, bfA, 0);
  gemm(bfA, omlp_w1t, omlp_b1, yv, Gg, 128, 128, 1);
  bn(yv, Gg, 128, omlp_bn2_g, omlp_bn2_b, vt, nullptr, 0);
  head_k<<<(Gg + 7) / 8, 256, 0, stream>>>(vt, omlp_w2, omlp_b2, yout);
}
